// Discriminator_16509854285849
// MI455X (gfx1250) — matrix operand order repaired and checked by reading, not compiled, not run on hardware
//
#include <hip/hip_runtime.h>

// ---------------------------------------------------------------------------
// Discriminator forward for MI455X (gfx1250, wave32, WMMA).
// All large matmuls run through v_wmma_f32_16x16x32_bf16 (bf16 in, f32 accum),
// with 2x2 register blocking (32x32 macro-tile per wave -> 4 WMMAs per 2
// A-fragment + 2 B-fragment loads, doubling arithmetic intensity vs 1x1).
// Softmax / bias / leaky-relu epilogues stay in f32.
// ---------------------------------------------------------------------------

#define C_DIM   512
#define NTOK    1024
#define HEADS   8
#define LAYERS  6
#define LEAKY_S 0.2f

typedef __attribute__((ext_vector_type(16))) __bf16 v16bf;
typedef __attribute__((ext_vector_type(8)))  float  v8f;

__device__ __forceinline__ unsigned short f2bf(float f) {
  union { float f; unsigned int u; } v; v.f = f;
  unsigned int u = v.u;
  unsigned int r = u + 0x7FFFu + ((u >> 16) & 1u);     // round-to-nearest-even
  if ((u & 0x7F800000u) == 0x7F800000u) r = u;         // Inf/NaN: truncate
  return (unsigned short)(r >> 16);
}

// ---------------------------------------------------------------------------
// Generic bf16 WMMA GEMM:  C[M,N] = act(alpha * A[M,K] @ Bt[N,K]^T + bias (+resid))
//   A  : bf16 row-major, lda (multiple of 32); row capacity padded to mult 32
//   Bt : bf16 row-major [N,K] (B pre-transposed), ldb (mult of 32), rows pad32
//   Outputs: optional f32 (ldcf) and/or bf16 (ldcb).
//   biasMode: 0 none, 1 per-col bias[col], 2 per-row bias[row]
//   actMode : 0 none, 1 leaky_relu(0.2)
// One wave computes one 32x32 macro-tile (2x2 of 16x16); 8 waves / 256-thr block.
// Fragment layouts follow CDNA5 ISA 7.12.2:
//   A (16x32 bf16): lane<16 holds row=lane, K = {0..7, 16..23}; lane>=16: K={8..15, 24..31}
//   B (32x16 bf16): lane<16 holds col=lane, K = {0..7, 16..23}; lane>=16: K={8..15, 24..31}
//   C (16x16 f32):  lane<16: col=lane, rows e; lane>=16: col=lane-16, rows 8+e
// ---------------------------------------------------------------------------
__global__ __launch_bounds__(256)
void k_gemm_bf16(const unsigned short* __restrict__ A, int lda,
                 const unsigned short* __restrict__ Bt, int ldb,
                 float* __restrict__ Cf, int ldcf,
                 unsigned short* __restrict__ Cb, int ldcb,
                 const float* __restrict__ bias, int biasMode,
                 const float* __restrict__ resid, int ldr,
                 int M, int N, int K, int actMode, float alpha)
{
  const int lane = threadIdx.x & 31;
  const int wave = threadIdx.x >> 5;
  const int nTn = (N + 31) >> 5;                 // 32-wide macro tiles
  const int nTm = (M + 31) >> 5;
  const int tile = blockIdx.x * 8 + wave;
  if (tile >= nTm * nTn) return;
  const int tm = tile / nTn;
  const int tn = tile % nTn;

  const int r16 = lane & 15;
  const int hi  = lane >> 4;                     // 0 for lanes 0-15, 1 for 16-31
  const unsigned short* aRow0 = A  + (size_t)(tm * 32 + r16) * lda + hi * 8;
  const unsigned short* aRow1 = aRow0 + (size_t)16 * lda;
  const unsigned short* bRow0 = Bt + (size_t)(tn * 32 + r16) * ldb + hi * 8;
  const unsigned short* bRow1 = bRow0 + (size_t)16 * ldb;

  v8f acc00 = {0.f,0.f,0.f,0.f,0.f,0.f,0.f,0.f};
  v8f acc01 = acc00, acc10 = acc00, acc11 = acc00;

  for (int kt = 0; kt < K; kt += 32) {
    union { v16bf v; uint4 q[2]; } fa0, fa1, fb0, fb1;
    fa0.q[0] = *(const uint4*)(aRow0 + kt);       // K chunk {0..7}+hi*8
    fa0.q[1] = *(const uint4*)(aRow0 + kt + 16);  // K chunk {16..23}+hi*8
    fa1.q[0] = *(const uint4*)(aRow1 + kt);
    fa1.q[1] = *(const uint4*)(aRow1 + kt + 16);
    fb0.q[0] = *(const uint4*)(bRow0 + kt);       // K {0..7}+hi*8
    fb0.q[1] = *(const uint4*)(bRow0 + kt + 16);  // K {16..23}+hi*8
    fb1.q[0] = *(const uint4*)(bRow1 + kt);
    fb1.q[1] = *(const uint4*)(bRow1 + kt + 16);
    if (kt + 32 < K) {
      __builtin_prefetch((const void*)(aRow0 + kt + 32), 0, 1);
      __builtin_prefetch((const void*)(aRow1 + kt + 32), 0, 1);
      __builtin_prefetch((const void*)(bRow0 + kt + 32), 0, 1);
      __builtin_prefetch((const void*)(bRow1 + kt + 32), 0, 1);
    }
    acc00 = __builtin_amdgcn_wmma_f32_16x16x32_bf16(false, fa0.v, false, fb0.v, (short)0, acc00, false, false);
    acc01 = __builtin_amdgcn_wmma_f32_16x16x32_bf16(false, fa0.v, false, fb1.v, (short)0, acc01, false, false);
    acc10 = __builtin_amdgcn_wmma_f32_16x16x32_bf16(false, fa1.v, false, fb0.v, (short)0, acc10, false, false);
    acc11 = __builtin_amdgcn_wmma_f32_16x16x32_bf16(false, fa1.v, false, fb1.v, (short)0, acc11, false, false);
  }

  v8f accs[2][2] = {{acc00, acc01}, {acc10, acc11}};
#pragma unroll
  for (int sn = 0; sn < 2; ++sn) {
    const int col = tn * 32 + sn * 16 + r16;
    if (col >= N) continue;
    const float bcol = (biasMode == 1 && bias) ? bias[col] : 0.f;
#pragma unroll
    for (int sm = 0; sm < 2; ++sm) {
      const int rowBase = tm * 32 + sm * 16 + hi * 8;
#pragma unroll
      for (int e = 0; e < 8; ++e) {
        const int row = rowBase + e;
        if (row >= M) continue;
        float v = accs[sm][sn][e] * alpha;
        if (biasMode == 1)              v += bcol;
        else if (biasMode == 2 && bias) v += bias[row];
        if (resid)                      v += resid[(size_t)row * ldr + col];
        if (actMode == 1)               v = v > 0.f ? v : v * LEAKY_S;
        if (Cf) Cf[(size_t)row * ldcf + col] = v;
        if (Cb) Cb[(size_t)row * ldcb + col] = f2bf(v);
      }
    }
  }
}

// W[K,N] f32 row-major  ->  Bt[Np,Kp] bf16 (transposed + zero padded), Np mult 32
__global__ void k_prep_wt(const float* __restrict__ W,
                          unsigned short* __restrict__ Bt,
                          int K, int N, int Kp, int Np)
{
  int idx = blockIdx.x * blockDim.x + threadIdx.x;
  int total = Np * Kp;
  if (idx >= total) return;
  int n = idx / Kp, k = idx % Kp;
  float v = (n < N && k < K) ? W[(size_t)k * N + n] : 0.f;
  Bt[idx] = f2bf(v);
}

// f32 [R,Cc] (lds) -> bf16 [Rp,Cp] zero padded
__global__ void k_cvt_pad(const float* __restrict__ src, int lds,
                          unsigned short* __restrict__ dst,
                          int R, int Cc, int Rp, int Cp)
{
  int idx = blockIdx.x * blockDim.x + threadIdx.x;
  int total = Rp * Cp;
  if (idx >= total) return;
  int r = idx / Cp, c = idx % Cp;
  dst[idx] = (r < R && c < Cc) ? f2bf(src[(size_t)r * lds + c]) : (unsigned short)0;
}

// copy f32 block into bf16 destination at column offset; broadcastRow=1 means
// src is a single [Cc] vector replicated over all rows.
__global__ void k_copy_block(const float* __restrict__ src, int lds, int broadcastRow,
                             unsigned short* __restrict__ dst, int ldd, int colOff,
                             int R, int Cc)
{
  int idx = blockIdx.x * blockDim.x + threadIdx.x;
  int total = R * Cc;
  if (idx >= total) return;
  int r = idx / Cc, c = idx % Cc;
  float v = broadcastRow ? src[c] : src[(size_t)r * lds + c];
  dst[(size_t)r * ldd + colOff + c] = f2bf(v);
}

// row-wise softmax (f32 in), one 256-thread block per row; optional bf16/f32 out
__global__ __launch_bounds__(256)
void k_softmax_rows(const float* __restrict__ S, int lds, int L,
                    unsigned short* __restrict__ Pb, float* __restrict__ Pf, int ldo)
{
  __shared__ float red[256];
  const int row = blockIdx.x;
  const int t = threadIdx.x;
  const float* s = S + (size_t)row * lds;
  float m = -3.4e38f;
  for (int i = t; i < L; i += 256) m = fmaxf(m, s[i]);
  red[t] = m; __syncthreads();
  for (int w = 128; w > 0; w >>= 1) { if (t < w) red[t] = fmaxf(red[t], red[t + w]); __syncthreads(); }
  m = red[0]; __syncthreads();
  float sum = 0.f;
  for (int i = t; i < L; i += 256) sum += __expf(s[i] - m);
  red[t] = sum; __syncthreads();
  for (int w = 128; w > 0; w >>= 1) { if (t < w) red[t] += red[t + w]; __syncthreads(); }
  const float inv = 1.f / red[0];
  for (int i = t; i < L; i += 256) {
    float p = __expf(s[i] - m) * inv;
    if (Pb) Pb[(size_t)row * ldo + i] = f2bf(p);
    if (Pf) Pf[(size_t)row * ldo + i] = p;
  }
}

// ---- attention-cluster pooling (collapsed algebra, f32, tiny) --------------
__global__ void k_cl_matvec(const float* __restrict__ Wk, const float* __restrict__ q,
                            float* __restrict__ w, int H, int Cd)
{
  int idx = blockIdx.x * blockDim.x + threadIdx.x;
  if (idx >= H * Cd) return;
  int hh = idx / Cd, c = idx % Cd;
  const float* Wh = Wk + ((size_t)hh * Cd + c) * Cd;
  const float* qh = q + (size_t)hh * Cd;
  float s = 0.f;
  for (int d = 0; d < Cd; ++d) s += Wh[d] * qh[d];
  w[idx] = s;
}
__global__ void k_cl_scores(const float* __restrict__ X, const float* __restrict__ w,
                            float* __restrict__ sc, int H, int Nt, int Cd, float scale)
{
  int idx = blockIdx.x * blockDim.x + threadIdx.x;
  if (idx >= H * Nt) return;
  int hh = idx / Nt, n = idx % Nt;
  const float* xr = X + (size_t)n * Cd;
  const float* wh = w + (size_t)hh * Cd;
  float s = 0.f;
  for (int c = 0; c < Cd; ++c) s += xr[c] * wh[c];
  sc[idx] = s * scale;
}
__global__ void k_cl_pool(const float* __restrict__ X, const float* __restrict__ a,
                          float* __restrict__ p, int H, int Nt, int Cd)
{
  int idx = blockIdx.x * blockDim.x + threadIdx.x;
  if (idx >= H * Cd) return;
  int hh = idx / Cd, c = idx % Cd;
  const float* ah = a + (size_t)hh * Nt;
  float s = 0.f;
  for (int n = 0; n < Nt; ++n) s += ah[n] * X[(size_t)n * Cd + c];
  p[idx] = s;
}
__global__ void k_cl_vproj(const float* __restrict__ p, const float* __restrict__ Wv,
                           const float* __restrict__ bv, float* __restrict__ out,
                           int H, int Cd)
{
  int idx = blockIdx.x * blockDim.x + threadIdx.x;
  if (idx >= H * Cd) return;
  int hh = idx / Cd, d = idx % Cd;
  const float* ph = p + (size_t)hh * Cd;
  const float* Wh = Wv + (size_t)hh * Cd * Cd;
  float s = bv[(size_t)hh * Cd + d];
  for (int c = 0; c < Cd; ++c) s += ph[c] * Wh[(size_t)c * Cd + d];
  out[idx] = s;
}

// ---------------------------------------------------------------------------
// Host-side launch helpers
// ---------------------------------------------------------------------------
static inline void gemm(hipStream_t st, const unsigned short* A, int lda,
                        const unsigned short* Bt, int ldb,
                        float* Cf, int ldcf, unsigned short* Cb, int ldcb,
                        const float* bias, int biasMode, const float* resid, int ldr,
                        int M, int N, int K, int act, float alpha)
{
  int tiles = ((M + 31) / 32) * ((N + 31) / 32);
  int blocks = (tiles + 7) / 8;
  k_gemm_bf16<<<blocks, 256, 0, st>>>(A, lda, Bt, ldb, Cf, ldcf, Cb, ldcb,
                                      bias, biasMode, resid, ldr, M, N, K, act, alpha);
}
static inline void prepw(hipStream_t st, const float* W, unsigned short* Bt, int K, int N)
{
  int Kp = (K + 31) & ~31, Np = (N + 31) & ~31;   // rows pad32 for macro-tile reads
  int total = Np * Kp;
  k_prep_wt<<<(total + 255) / 256, 256, 0, st>>>(W, Bt, K, N, Kp, Np);
}
static inline void cvt(hipStream_t st, const float* src, int lds, unsigned short* dst,
                       int R, int Cc, int Rp, int Cp)
{
  int total = Rp * Cp;
  k_cvt_pad<<<(total + 255) / 256, 256, 0, st>>>(src, lds, dst, R, Cc, Rp, Cp);
}
static inline void copyb(hipStream_t st, const float* src, int lds, int bro,
                         unsigned short* dst, int ldd, int colOff, int R, int Cc)
{
  int total = R * Cc;
  k_copy_block<<<(total + 255) / 256, 256, 0, st>>>(src, lds, bro, dst, ldd, colOff, R, Cc);
}

extern "C" void kernel_launch(void* const* d_in, const int* in_sizes, int n_in,
                              void* d_out, int out_size, void* d_ws, size_t ws_size,
                              hipStream_t stream)
{
  (void)in_sizes; (void)n_in; (void)out_size; (void)ws_size;
  const float SCALE = 0.044194173824159216f; // 1/sqrt(512)

  // ---- input mapping -------------------------------------------------------
  // Assumed flattening: top-level dict insertion order (x, l, params); params
  // leaves in jax tree order (dict keys sorted, lists/tuples in order).
  const float* x = (const float*)d_in[0];          // [1024,10]
  const float* l = (const float*)d_in[1];          // [1,1]
  int p = 2;
  const float *ajW[3], *ajB[3];
  for (int i = 0; i < 3; ++i) { ajW[i] = (const float*)d_in[p++]; ajB[i] = (const float*)d_in[p++]; }
  struct LayerP { const float *Wfc,*Wk,*Wq,*Wv,*bfc,*bk,*bq,*bv,*mW0,*mb0,*mW1,*mb1; } Lp[LAYERS];
  for (int i = 0; i < LAYERS; ++i) {               // mha keys sorted: Wfc,Wk,Wq,Wv,bfc,bk,bq,bv
    Lp[i].Wfc = (const float*)d_in[p++]; Lp[i].Wk  = (const float*)d_in[p++];
    Lp[i].Wq  = (const float*)d_in[p++]; Lp[i].Wv  = (const float*)d_in[p++];
    Lp[i].bfc = (const float*)d_in[p++]; Lp[i].bk  = (const float*)d_in[p++];
    Lp[i].bq  = (const float*)d_in[p++]; Lp[i].bv  = (const float*)d_in[p++];
    Lp[i].mW0 = (const float*)d_in[p++]; Lp[i].mb0 = (const float*)d_in[p++];
    Lp[i].mW1 = (const float*)d_in[p++]; Lp[i].mb1 = (const float*)d_in[p++];
  }
  const float *dpW[2], *dpB[2];
  for (int i = 0; i < 2; ++i) { dpW[i] = (const float*)d_in[p++]; dpB[i] = (const float*)d_in[p++]; }
  const float* fW = (const float*)d_in[p++];
  const float* fb = (const float*)d_in[p++];
  const float *ffW[3], *ffB[3];
  for (int i = 0; i < 3; ++i) { ffW[i] = (const float*)d_in[p++]; ffB[i] = (const float*)d_in[p++]; }
  const float *lnW[2], *lnB[2];
  for (int i = 0; i < 2; ++i) { lnW[i] = (const float*)d_in[p++]; lnB[i] = (const float*)d_in[p++]; }
  struct StageP { const float *Wk,*Wv,*bk,*bv,*q,*mW0,*mb0,*mW1,*mb1; } Sp[3];
  for (int i = 0; i < 3; ++i) {                    // cluster keys sorted: Wk,Wv,bk,bv,q
    Sp[i].Wk  = (const float*)d_in[p++]; Sp[i].Wv  = (const float*)d_in[p++];
    Sp[i].bk  = (const float*)d_in[p++]; Sp[i].bv  = (const float*)d_in[p++];
    Sp[i].q   = (const float*)d_in[p++];
    Sp[i].mW0 = (const float*)d_in[p++]; Sp[i].mb0 = (const float*)d_in[p++];
    Sp[i].mW1 = (const float*)d_in[p++]; Sp[i].mb1 = (const float*)d_in[p++];
  }

  // ---- workspace layout ----------------------------------------------------
  char* W8 = (char*)d_ws;
  size_t off = 0;
  auto al = [&](size_t bytes) -> size_t { size_t o = off; off += (bytes + 255) & ~(size_t)255; return o; };
  const size_t oBFX  = al((size_t)NTOK * 32 * 2);          // x bf16 padded to K=32
  const size_t oFAJ  = al((size_t)NTOK * C_DIM * 4);       // atom_judge out f32
  const size_t oFLV  = al((size_t)C_DIM * 4);              // length vector f32
  const size_t oFH   = al((size_t)NTOK * C_DIM * 4);       // current h f32
  const size_t oBFH  = al((size_t)NTOK * C_DIM * 2);       // current h bf16
  const size_t oFT1  = al((size_t)NTOK * C_DIM * 4);
  const size_t oBT1  = al((size_t)NTOK * C_DIM * 2);
  const size_t oBT2  = al((size_t)NTOK * C_DIM * 2);
  const size_t oBQ   = al((size_t)NTOK * C_DIM * 2);
  const size_t oBK   = al((size_t)NTOK * C_DIM * 2);
  const size_t oBVT  = al((size_t)C_DIM * NTOK * 2);       // V^T bf16 [512,1024]
  const size_t oFS   = al((size_t)NTOK * NTOK * 4);        // scores f32
  const size_t oBP   = al((size_t)NTOK * NTOK * 2);        // probs bf16
  const size_t oBO   = al((size_t)NTOK * (HEADS * C_DIM) * 2); // head-concat bf16
  const size_t oBCAT = al((size_t)NTOK * (2 * C_DIM) * 2); // concat buffer bf16
  const size_t oWA   = al((size_t)C_DIM * C_DIM * 2);      // weight slabs (recycled)
  const size_t oWB   = al((size_t)C_DIM * C_DIM * 2);
  const size_t oWC   = al((size_t)C_DIM * C_DIM * 2);
  const size_t oWFC  = al((size_t)C_DIM * (HEADS * C_DIM) * 2);
  const size_t oWD   = al((size_t)C_DIM * (2 * C_DIM) * 2);
  const size_t oBFL  = al((size_t)32 * 32 * 2);            // length input bf16 (32-row cap)
  const size_t oBLH  = al((size_t)32 * C_DIM * 2);         // length hidden bf16 (32-row cap)
  const size_t oFW   = al((size_t)16 * C_DIM * 4);
  const size_t oFSC  = al((size_t)16 * NTOK * 4);
  const size_t oFSA  = al((size_t)16 * NTOK * 4);
  const size_t oFP   = al((size_t)16 * C_DIM * 4);
  const size_t oFCO  = al((size_t)16 * C_DIM * 4);
  const size_t oBR   = al((size_t)32 * C_DIM * 2);         // reduce input bf16 (32-row cap)
  const size_t oFR2  = al((size_t)16 * C_DIM * 4);

  unsigned short* BFX  = (unsigned short*)(W8 + oBFX);
  float*          FAJ  = (float*)(W8 + oFAJ);
  float*          FLV  = (float*)(W8 + oFLV);
  float*          FH   = (float*)(W8 + oFH);
  unsigned short* BFH  = (unsigned short*)(W8 + oBFH);
  float*          FT1  = (float*)(W8 + oFT1);
  unsigned short* BT1  = (unsigned short*)(W8 + oBT1);
  unsigned short* BT2  = (unsigned short*)(W8 + oBT2);
  unsigned short* BQ   = (unsigned short*)(W8 + oBQ);
  unsigned short* BK   = (unsigned short*)(W8 + oBK);
  unsigned short* BVT  = (unsigned short*)(W8 + oBVT);
  float*          FS   = (float*)(W8 + oFS);
  unsigned short* BP   = (unsigned short*)(W8 + oBP);
  unsigned short* BO   = (unsigned short*)(W8 + oBO);
  unsigned short* BCAT = (unsigned short*)(W8 + oBCAT);
  unsigned short* WA   = (unsigned short*)(W8 + oWA);
  unsigned short* WB   = (unsigned short*)(W8 + oWB);
  unsigned short* WC   = (unsigned short*)(W8 + oWC);
  unsigned short* WFC  = (unsigned short*)(W8 + oWFC);
  unsigned short* WD   = (unsigned short*)(W8 + oWD);
  unsigned short* BFL  = (unsigned short*)(W8 + oBFL);
  unsigned short* BLH  = (unsigned short*)(W8 + oBLH);
  float*          FWv  = (float*)(W8 + oFW);
  float*          FSC  = (float*)(W8 + oFSC);
  float*          FSA  = (float*)(W8 + oFSA);
  float*          FP   = (float*)(W8 + oFP);
  float*          FCO  = (float*)(W8 + oFCO);
  unsigned short* BR   = (unsigned short*)(W8 + oBR);
  float*          FR2  = (float*)(W8 + oFR2);

  // ---- 1. inputs to bf16 ---------------------------------------------------
  cvt(stream, x, 10, BFX, NTOK, 10, NTOK, 32);

  // ---- 2. atom_judge MLP [10,512,512,512] ---------------------------------
  prepw(stream, ajW[0], WA, 10, C_DIM);
  gemm(stream, BFX, 32, WA, 32, nullptr, 0, BT1, C_DIM, ajB[0], 1, nullptr, 0, NTOK, C_DIM, 32, 1, 1.f);
  prepw(stream, ajW[1], WB, C_DIM, C_DIM);
  gemm(stream, BT1, C_DIM, WB, C_DIM, nullptr, 0, BT2, C_DIM, ajB[1], 1, nullptr, 0, NTOK, C_DIM, C_DIM, 1, 1.f);
  prepw(stream, ajW[2], WA, C_DIM, C_DIM);
  gemm(stream, BT2, C_DIM, WA, C_DIM, FAJ, C_DIM, nullptr, 0, ajB[2], 1, nullptr, 0, NTOK, C_DIM, C_DIM, 0, 1.f);

  // ---- 3. length MLP [1,512,512] ------------------------------------------
  cvt(stream, l, 1, BFL, 1, 1, 32, 32);
  prepw(stream, lnW[0], WA, 1, C_DIM);
  gemm(stream, BFL, 32, WA, 32, nullptr, 0, BLH, C_DIM, lnB[0], 1, nullptr, 0, 1, C_DIM, 32, 1, 1.f);
  prepw(stream, lnW[1], WB, C_DIM, C_DIM);
  gemm(stream, BLH, C_DIM, WB, C_DIM, FLV, C_DIM, nullptr, 0, lnB[1], 1, nullptr, 0, 1, C_DIM, C_DIM, 0, 1.f);

  // ---- 4. dense_pre [10,512,512] ------------------------------------------
  prepw(stream, dpW[0], WA, 10, C_DIM);
  gemm(stream, BFX, 32, WA, 32, nullptr, 0, BT1, C_DIM, dpB[0], 1, nullptr, 0, NTOK, C_DIM, 32, 1, 1.f);
  prepw(stream, dpW[1], WB, C_DIM, C_DIM);
  gemm(stream, BT1, C_DIM, WB, C_DIM, FH, C_DIM, BFH, C_DIM, dpB[1], 1, nullptr, 0, NTOK, C_DIM, C_DIM, 0, 1.f);

  // ---- 5. attention stack --------------------------------------------------
  for (int li = 0; li < LAYERS; ++li) {
    const LayerP& L = Lp[li];
    for (int hh = 0; hh < HEADS; ++hh) {
      const size_t ws = (size_t)hh * C_DIM * C_DIM;
      // Q, K projections (bf16 out)
      prepw(stream, L.Wq + ws, WA, C_DIM, C_DIM);
      gemm(stream, BFH, C_DIM, WA, C_DIM, nullptr, 0, BQ, C_DIM, L.bq + hh * C_DIM, 1, nullptr, 0, NTOK, C_DIM, C_DIM, 0, 1.f);
      prepw(stream, L.Wk + ws, WB, C_DIM, C_DIM);
      gemm(stream, BFH, C_DIM, WB, C_DIM, nullptr, 0, BK, C_DIM, L.bk + hh * C_DIM, 1, nullptr, 0, NTOK, C_DIM, C_DIM, 0, 1.f);
      // V^T directly: Vt[d,n] = sum_c Wv[c,d] h[n,c]; A = Wv^T (prepped), Bt = h
      prepw(stream, L.Wv + ws, WC, C_DIM, C_DIM);
      gemm(stream, WC, C_DIM, BFH, C_DIM, nullptr, 0, BVT, NTOK, L.bv + hh * C_DIM, 2, nullptr, 0, C_DIM, NTOK, C_DIM, 0, 1.f);
      // scores = scale * Q K^T  (Bt = K row-major [N,C] already)
      gemm(stream, BQ, C_DIM, BK, C_DIM, FS, NTOK, nullptr, 0, nullptr, 0, nullptr, 0, NTOK, NTOK, C_DIM, 0, SCALE);
      k_softmax_rows<<<NTOK, 256, 0, stream>>>(FS, NTOK, NTOK, BP, nullptr, NTOK);
      // O_h = P V  (Bt = V^T [512,1024]); write bf16 straight into head-concat slot
      gemm(stream, BP, NTOK, BVT, NTOK, nullptr, 0, BO + hh * C_DIM, HEADS * C_DIM,
           nullptr, 0, nullptr, 0, NTOK, C_DIM, NTOK, 0, 1.f);
    }
    // fc + residual
    prepw(stream, L.Wfc, WFC, HEADS * C_DIM, C_DIM);
    gemm(stream, BO, HEADS * C_DIM, WFC, HEADS * C_DIM, FT1, C_DIM, BT1, C_DIM,
         L.bfc, 1, FH, C_DIM, NTOK, C_DIM, HEADS * C_DIM, 0, 1.f);
    // layer MLP [512,512,512]
    prepw(stream, L.mW0, WA, C_DIM, C_DIM);
    gemm(stream, BT1, C_DIM, WA, C_DIM, nullptr, 0, BT2, C_DIM, L.mb0, 1, nullptr, 0, NTOK, C_DIM, C_DIM, 1, 1.f);
    prepw(stream, L.mW1, WB, C_DIM, C_DIM);
    gemm(stream, BT2, C_DIM, WB, C_DIM, FH, C_DIM, BFH, C_DIM, L.mb1, 1, nullptr, 0, NTOK, C_DIM, C_DIM, 0, 1.f);
  }

  // ---- 6. final: concat([h, lv]) @ Wf + bf --------------------------------
  copyb(stream, FH,  C_DIM, 0, BCAT, 2 * C_DIM, 0,     NTOK, C_DIM);
  copyb(stream, FLV, C_DIM, 1, BCAT, 2 * C_DIM, C_DIM, NTOK, C_DIM);
  prepw(stream, fW, WD, 2 * C_DIM, C_DIM);
  gemm(stream, BCAT, 2 * C_DIM, WD, 2 * C_DIM, FT1, C_DIM, nullptr, 0, fb, 1, nullptr, 0, NTOK, C_DIM, 2 * C_DIM, 0, 1.f);

  // ---- 7. final_final: MLP(concat([h, aj])) [1024,512,512,512] ------------
  copyb(stream, FT1, C_DIM, 0, BCAT, 2 * C_DIM, 0,     NTOK, C_DIM);
  copyb(stream, FAJ, C_DIM, 0, BCAT, 2 * C_DIM, C_DIM, NTOK, C_DIM);
  prepw(stream, ffW[0], WD, 2 * C_DIM, C_DIM);
  gemm(stream, BCAT, 2 * C_DIM, WD, 2 * C_DIM, nullptr, 0, BT1, C_DIM, ffB[0], 1, nullptr, 0, NTOK, C_DIM, 2 * C_DIM, 1, 1.f);
  prepw(stream, ffW[1], WA, C_DIM, C_DIM);
  gemm(stream, BT1, C_DIM, WA, C_DIM, nullptr, 0, BT2, C_DIM, ffB[1], 1, nullptr, 0, NTOK, C_DIM, C_DIM, 1, 1.f);
  prepw(stream, ffW[2], WB, C_DIM, C_DIM);
  gemm(stream, BT2, C_DIM, WB, C_DIM, FH, C_DIM, nullptr, 0, ffB[2], 1, nullptr, 0, NTOK, C_DIM, C_DIM, 0, 1.f);

  // ---- 8. reduce: 16 -> 4 -> 1 attention-pooling clusters ------------------
  // Collapsed: scores_n = (h_n . (Wk q)) * scale (q.bk shift is softmax-invariant)
  //            out = (sum_n a_n h_n) Wv + bv      (since sum_n a_n == 1)
  const float* X = FH;
  int Ntok = NTOK;
  const int hcs[3] = {16, 4, 1};
  for (int s = 0; s < 3; ++s) {
    const StageP& S = Sp[s];
    const int hc = hcs[s];
    k_cl_matvec<<<(hc * C_DIM + 255) / 256, 256, 0, stream>>>(S.Wk, S.q, FWv, hc, C_DIM);
    k_cl_scores<<<(hc * Ntok + 255) / 256, 256, 0, stream>>>(X, FWv, FSC, hc, Ntok, C_DIM, SCALE);
    k_softmax_rows<<<hc, 256, 0, stream>>>(FSC, Ntok, Ntok, nullptr, FSA, Ntok);
    k_cl_pool<<<(hc * C_DIM + 255) / 256, 256, 0, stream>>>(X, FSA, FP, hc, Ntok, C_DIM);
    k_cl_vproj<<<(hc * C_DIM + 255) / 256, 256, 0, stream>>>(FP, S.Wv, S.bv, FCO, hc, C_DIM);
    // stage MLP: [512,512,512] (stages 0,1) or [512,512,1] (stage 2)
    cvt(stream, FCO, C_DIM, BR, hc, C_DIM, 32, C_DIM);
    prepw(stream, S.mW0, WA, C_DIM, C_DIM);
    gemm(stream, BR, C_DIM, WA, C_DIM, nullptr, 0, BT2, C_DIM, S.mb0, 1, nullptr, 0, hc, C_DIM, C_DIM, 1, 1.f);
    const int Nout = (s == 2) ? 1 : C_DIM;
    prepw(stream, S.mW1, WB, C_DIM, Nout);
    if (s < 2) {
      gemm(stream, BT2, C_DIM, WB, C_DIM, FR2, C_DIM, nullptr, 0, S.mb1, 1, nullptr, 0, hc, Nout, C_DIM, 0, 1.f);
      X = FR2; Ntok = hc;
    } else {
      gemm(stream, BT2, C_DIM, WB, C_DIM, (float*)d_out, 1, nullptr, 0, S.mb1, 1, nullptr, 0, 1, 1, C_DIM, 0, 1.f);
    }
  }
}
